// LocalGlobalSelfAttention_16149077033187
// MI455X (gfx1250) — matrix operand order repaired and checked by reading, not compiled, not run on hardware
//
#include <hip/hip_runtime.h>

// ---------------------------------------------------------------------------
// Types for gfx1250 WMMA
// ---------------------------------------------------------------------------
typedef __bf16 bf16_t;
typedef __attribute__((ext_vector_type(16))) __bf16 v16bf;
typedef __attribute__((ext_vector_type(8)))  float  v8f;
typedef __attribute__((ext_vector_type(4)))  int    v4i;

#define WMMA_BF16(a, b, c) \
  __builtin_amdgcn_wmma_f32_16x16x32_bf16(false, (a), false, (b), (short)0, (c), false, false)

// gfx1250 async global->LDS path (guarded; falls back to plain vector copy)
#if defined(__has_builtin)
# if __has_builtin(__builtin_amdgcn_global_load_async_to_lds_b128) && \
     __has_builtin(__builtin_amdgcn_s_wait_asynccnt)
#  define HAVE_ASYNC 1
# endif
#endif
#ifndef HAVE_ASYNC
# define HAVE_ASYNC 0
#endif

#define AS_GLOBAL __attribute__((address_space(1)))
#define AS_LDS    __attribute__((address_space(3)))

// Problem constants
#define BB 4
#define SS 2048
#define HH 16
#define NW 8
#define WW (SS / NW)        // 256
#define MTOT (BB * SS)      // 8192

#define AP 40               // padded LDS row stride for 32-wide tiles (80 B)
#define KP 72               // padded LDS row stride for 64-wide tiles (144 B)

// float -> bf16 (round-to-nearest-even) without relying on __bf16 casts
__device__ __forceinline__ bf16_t f2bf(float f) {
  union { float f; unsigned u; } a; a.f = f;
  unsigned r = (a.u + 0x7FFFu + ((a.u >> 16) & 1u)) >> 16;
  union { unsigned short s; bf16_t b; } c; c.s = (unsigned short)r;
  return c.b;
}

// copy 16 B global -> LDS (async on gfx1250 when available)
__device__ __forceinline__ void cp16_g2lds(bf16_t* lds_dst, const bf16_t* g_src) {
#if HAVE_ASYNC
  __builtin_amdgcn_global_load_async_to_lds_b128(
      (AS_GLOBAL v4i*)(bf16_t*)g_src,
      (AS_LDS v4i*)lds_dst, 0, 0);
#else
  *reinterpret_cast<uint4*>(lds_dst) = *reinterpret_cast<const uint4*>(g_src);
#endif
}
__device__ __forceinline__ void async_join() {
#if HAVE_ASYNC
  __builtin_amdgcn_s_wait_asynccnt(0);
#endif
}

// Fragment builders: two aligned 16B LDS loads -> v16bf, no repacking.
union Frag { uint4 q[2]; v16bf v; };

// A-layout (16x32): element i -> K = 8*half + i  (i<8), 16 + 8*half + (i-8)
__device__ __forceinline__ v16bf ldsA_frag(const bf16_t* row, int half) {
  Frag f;
  f.q[0] = *reinterpret_cast<const uint4*>(row + 8 * half);
  f.q[1] = *reinterpret_cast<const uint4*>(row + 16 + 8 * half);
  return f.v;
}
// B-layout (32x16): element i -> K = 8*half + i (i<8), 16 + 8*half + (i-8) ; "row" = the lane's N-row
// in a K-contiguous (transposed) LDS tile.
__device__ __forceinline__ v16bf ldsB_frag(const bf16_t* row, int half) {
  Frag f;
  f.q[0] = *reinterpret_cast<const uint4*>(row + 8 * half);
  f.q[1] = *reinterpret_cast<const uint4*>(row + 16 + 8 * half);
  return f.v;
}

// ---------------------------------------------------------------------------
// fp32 -> bf16 conversion (grid-stride)
// ---------------------------------------------------------------------------
__global__ __launch_bounds__(256)
void cvt_bf16_kernel(const float* __restrict__ in, bf16_t* __restrict__ out, int n) {
  for (int i = blockIdx.x * blockDim.x + threadIdx.x; i < n; i += gridDim.x * blockDim.x)
    out[i] = f2bf(in[i]);
}

// ---------------------------------------------------------------------------
// GEMM: C[8192,1024](bf16) = A[8192,1024](bf16) x W[1024,1024](bf16) + bias[n]
// Block = 128 thr (4 waves), tile 128(M) x 64(N), K-step 32, 8 WMMA/wave/step.
// B tile staged TRANSPOSED so B-fragment loads are 2x ds_load_b128.
// ---------------------------------------------------------------------------
__global__ __launch_bounds__(128)
void gemm_qkv_kernel(const bf16_t* __restrict__ A, const bf16_t* __restrict__ W,
                     const float* __restrict__ bias, bf16_t* __restrict__ C) {
  __shared__ alignas(32) bf16_t As[128][AP];
  __shared__ alignas(32) bf16_t BsT[64][AP];
  const int tid  = threadIdx.x;
  const int wave = tid >> 5, lane = tid & 31;
  const int half = lane >> 4, ln = lane & 15;
  const int mbase = blockIdx.x * 128;
  const int nbase = blockIdx.y * 64;

  v8f acc[2][4] = {};
  for (int kb = 0; kb < 1024; kb += 32) {
    __syncthreads();
    // A tile: 128 rows x 32 cols = 512 x 16B chunks (async when available)
    for (int t = tid; t < 512; t += 128) {
      int row = t >> 2, seg = t & 3;
      cp16_g2lds(&As[row][seg * 8], A + (size_t)(mbase + row) * 1024 + kb + seg * 8);
    }
    // B tile: 32 rows x 64 cols, stored transposed BsT[n][k]
    for (int t = tid; t < 256; t += 128) {
      int row = t >> 3, seg = t & 7;
      uint4 d = *reinterpret_cast<const uint4*>(W + (size_t)(kb + row) * 1024 + nbase + seg * 8);
      const bf16_t* e = reinterpret_cast<const bf16_t*>(&d);
#pragma unroll
      for (int c = 0; c < 8; ++c) BsT[seg * 8 + c][row] = e[c];
    }
    async_join();
    __syncthreads();

    v16bf a0 = ldsA_frag(&As[32 * wave + ln][0], half);
    v16bf a1 = ldsA_frag(&As[32 * wave + 16 + ln][0], half);
#pragma unroll
    for (int j = 0; j < 4; ++j) {
      v16bf b = ldsB_frag(&BsT[16 * j + ln][0], half);
      acc[0][j] = WMMA_BF16(a0, b, acc[0][j]);
      acc[1][j] = WMMA_BF16(a1, b, acc[1][j]);
    }
  }
#pragma unroll
  for (int j = 0; j < 4; ++j) {
    int n = nbase + 16 * j + ln;
    float bv = bias[n];
#pragma unroll
    for (int u = 0; u < 2; ++u)
#pragma unroll
      for (int r = 0; r < 8; ++r) {
        size_t m = (size_t)(mbase + 32 * wave + 16 * u + r + 8 * half);
        C[m * 1024 + n] = f2bf(acc[u][j][r] + bv);
      }
  }
}

// ---------------------------------------------------------------------------
// Flash attention: block = (64 q rows) x (head) x (sequence); 4 waves.
// KV streamed in chunks of 32; V staged transposed; 8 WMMAs / chunk / wave.
// q,k,v,o layout: [row = seq*seq_len + s][h*64 + d] (bf16)
// ---------------------------------------------------------------------------
__global__ __launch_bounds__(128)
void attn_kernel(const bf16_t* __restrict__ Q, const bf16_t* __restrict__ Km,
                 const bf16_t* __restrict__ V, bf16_t* __restrict__ O, int seq_len) {
  __shared__ alignas(32) bf16_t Qs[64][KP];
  __shared__ alignas(32) bf16_t Ks[32][KP];
  __shared__ alignas(32) bf16_t VsT[64][AP];
  __shared__ alignas(32) bf16_t Ps[4][16][AP];

  const int tid  = threadIdx.x;
  const int wave = tid >> 5, lane = tid & 31;
  const int half = lane >> 4, ln = lane & 15;
  const int h     = blockIdx.y;
  const int seq   = blockIdx.z;
  const int qbase = blockIdx.x * 64;
  const size_t seq_row0 = (size_t)seq * seq_len;
  const int hk0 = h * 64;
  const float scale = 0.125f;  // 1/sqrt(64)

  // stage 64x64 Q tile (row-major, async)
  for (int t = tid; t < 512; t += 128) {
    int row = t >> 3, seg = t & 7;
    cp16_g2lds(&Qs[row][seg * 8], Q + (seq_row0 + qbase + row) * 1024 + hk0 + seg * 8);
  }
  async_join();
  __syncthreads();

  const bf16_t* qrow = &Qs[16 * wave + ln][0];
  v16bf aq0 = ldsA_frag(qrow, half);        // head dims [0,32)
  v16bf aq1 = ldsA_frag(qrow + 32, half);   // head dims [32,64)

  v8f o[4] = {};
  float row_max[8], row_sum[8];
#pragma unroll
  for (int r = 0; r < 8; ++r) { row_max[r] = -1e30f; row_sum[r] = 0.f; }

  for (int kv = 0; kv < seq_len; kv += 32) {
    __syncthreads();
    // K chunk row-major (async), V chunk transposed VsT[d][kv]
    for (int t = tid; t < 256; t += 128) {
      int row = t >> 3, seg = t & 7;
      cp16_g2lds(&Ks[row][seg * 8], Km + (seq_row0 + kv + row) * 1024 + hk0 + seg * 8);
    }
    for (int t = tid; t < 256; t += 128) {
      int row = t >> 3, seg = t & 7;
      uint4 d = *reinterpret_cast<const uint4*>(V + (seq_row0 + kv + row) * 1024 + hk0 + seg * 8);
      const bf16_t* e = reinterpret_cast<const bf16_t*>(&d);
#pragma unroll
      for (int c = 0; c < 8; ++c) VsT[seg * 8 + c][row] = e[c];
    }
    async_join();
    __syncthreads();

    // scores: c0 -> kv cols [0,16), c1 -> [16,32); contract over head dim 64.
    // B(kd, n) = K[n][kd] -> K rows are already K-contiguous.
    v8f c0 = {}, c1 = {};
#pragma unroll
    for (int t2 = 0; t2 < 2; ++t2) {
      v16bf b0 = ldsB_frag(&Ks[ln][32 * t2], half);
      v16bf b1 = ldsB_frag(&Ks[16 + ln][32 * t2], half);
      v16bf aq = t2 ? aq1 : aq0;
      c0 = WMMA_BF16(aq, b0, c0);
      c1 = WMMA_BF16(aq, b1, c1);
    }

    // online softmax, per C-fragment row r (row = r + 8*half)
#pragma unroll
    for (int r = 0; r < 8; ++r) {
      float s0 = c0[r] * scale, s1 = c1[r] * scale;
      float mx = fmaxf(s0, s1);
#pragma unroll
      for (int off = 1; off < 16; off <<= 1) mx = fmaxf(mx, __shfl_xor(mx, off, 32));
      float mnew  = fmaxf(row_max[r], mx);
      float alpha = __expf(row_max[r] - mnew);
      float p0 = __expf(s0 - mnew), p1 = __expf(s1 - mnew);
      float ps = p0 + p1;
#pragma unroll
      for (int off = 1; off < 16; off <<= 1) ps += __shfl_xor(ps, off, 32);
      row_sum[r] = row_sum[r] * alpha + ps;
      row_max[r] = mnew;
#pragma unroll
      for (int j = 0; j < 4; ++j) o[j][r] *= alpha;
      int ml = r + 8 * half;
      Ps[wave][ml][ln]      = f2bf(p0);
      Ps[wave][ml][16 + ln] = f2bf(p1);
    }

    // rebuild P as an A-fragment (per-wave LDS region; DS in-order per wave)
    v16bf ap = ldsA_frag(&Ps[wave][ln][0], half);
    // O += P * V : B(k=kv, n=d) = VsT[d][kv], K-contiguous rows.
#pragma unroll
    for (int j = 0; j < 4; ++j) {
      v16bf bv = ldsB_frag(&VsT[16 * j + ln][0], half);
      o[j] = WMMA_BF16(ap, bv, o[j]);
    }
  }

#pragma unroll
  for (int r = 0; r < 8; ++r) {
    float inv = 1.f / row_sum[r];
    size_t m = seq_row0 + qbase + 16 * wave + r + 8 * half;
#pragma unroll
    for (int j = 0; j < 4; ++j)
      O[m * 1024 + hk0 + 16 * j + ln] = f2bf(o[j][r] * inv);
  }
}

// ---------------------------------------------------------------------------
// Fused output projections + residual:
// Y = x + Og*Wg + Ol*Wl + bg + bl     (fp32 out), tile 128x64
// ---------------------------------------------------------------------------
__global__ __launch_bounds__(128)
void fuse_proj_kernel(const bf16_t* __restrict__ Og, const bf16_t* __restrict__ Wg,
                      const bf16_t* __restrict__ Ol, const bf16_t* __restrict__ Wl,
                      const float* __restrict__ x, const float* __restrict__ bg,
                      const float* __restrict__ bl, float* __restrict__ Y) {
  __shared__ alignas(32) bf16_t As1[128][AP];
  __shared__ alignas(32) bf16_t As2[128][AP];
  __shared__ alignas(32) bf16_t BsT1[64][AP];
  __shared__ alignas(32) bf16_t BsT2[64][AP];
  const int tid  = threadIdx.x;
  const int wave = tid >> 5, lane = tid & 31;
  const int half = lane >> 4, ln = lane & 15;
  const int mbase = blockIdx.x * 128;
  const int nbase = blockIdx.y * 64;

  v8f acc[2][4] = {};
  for (int kb = 0; kb < 1024; kb += 32) {
    __syncthreads();
    for (int t = tid; t < 512; t += 128) {
      int row = t >> 2, seg = t & 3;
      cp16_g2lds(&As1[row][seg * 8], Og + (size_t)(mbase + row) * 1024 + kb + seg * 8);
      cp16_g2lds(&As2[row][seg * 8], Ol + (size_t)(mbase + row) * 1024 + kb + seg * 8);
    }
    for (int t = tid; t < 256; t += 128) {
      int row = t >> 3, seg = t & 7;
      uint4 d1 = *reinterpret_cast<const uint4*>(Wg + (size_t)(kb + row) * 1024 + nbase + seg * 8);
      uint4 d2 = *reinterpret_cast<const uint4*>(Wl + (size_t)(kb + row) * 1024 + nbase + seg * 8);
      const bf16_t* e1 = reinterpret_cast<const bf16_t*>(&d1);
      const bf16_t* e2 = reinterpret_cast<const bf16_t*>(&d2);
#pragma unroll
      for (int c = 0; c < 8; ++c) {
        BsT1[seg * 8 + c][row] = e1[c];
        BsT2[seg * 8 + c][row] = e2[c];
      }
    }
    async_join();
    __syncthreads();

    v16bf a10 = ldsA_frag(&As1[32 * wave + ln][0], half);
    v16bf a11 = ldsA_frag(&As1[32 * wave + 16 + ln][0], half);
    v16bf a20 = ldsA_frag(&As2[32 * wave + ln][0], half);
    v16bf a21 = ldsA_frag(&As2[32 * wave + 16 + ln][0], half);
#pragma unroll
    for (int j = 0; j < 4; ++j) {
      v16bf b1 = ldsB_frag(&BsT1[16 * j + ln][0], half);
      v16bf b2 = ldsB_frag(&BsT2[16 * j + ln][0], half);
      acc[0][j] = WMMA_BF16(a10, b1, acc[0][j]);
      acc[0][j] = WMMA_BF16(a20, b2, acc[0][j]);
      acc[1][j] = WMMA_BF16(a11, b1, acc[1][j]);
      acc[1][j] = WMMA_BF16(a21, b2, acc[1][j]);
    }
  }
#pragma unroll
  for (int j = 0; j < 4; ++j) {
    int n = nbase + 16 * j + ln;
    float bv = bg[n] + bl[n];
#pragma unroll
    for (int u = 0; u < 2; ++u)
#pragma unroll
      for (int r = 0; r < 8; ++r) {
        size_t m = (size_t)(mbase + 32 * wave + 16 * u + r + 8 * half);
        Y[m * 1024 + n] = acc[u][j][r] + bv + x[m * 1024 + n];
      }
  }
}

// ---------------------------------------------------------------------------
// LayerNorm over last dim (1024), one block per row
// ---------------------------------------------------------------------------
__global__ __launch_bounds__(256)
void layernorm_kernel(const float* __restrict__ Y, const float* __restrict__ gamma,
                      const float* __restrict__ beta, float* __restrict__ out) {
  __shared__ float red[256];
  const int row = blockIdx.x, tid = threadIdx.x;
  const float* y = Y + (size_t)row * 1024;
  float v[4];
  float s = 0.f;
#pragma unroll
  for (int i = 0; i < 4; ++i) { v[i] = y[tid + 256 * i]; s += v[i]; }
  red[tid] = s; __syncthreads();
  for (int st = 128; st > 0; st >>= 1) { if (tid < st) red[tid] += red[tid + st]; __syncthreads(); }
  float mu = red[0] * (1.f / 1024.f);
  __syncthreads();
  float q = 0.f;
#pragma unroll
  for (int i = 0; i < 4; ++i) { float d = v[i] - mu; q += d * d; }
  red[tid] = q; __syncthreads();
  for (int st = 128; st > 0; st >>= 1) { if (tid < st) red[tid] += red[tid + st]; __syncthreads(); }
  float rinv = rsqrtf(red[0] * (1.f / 1024.f) + 1e-3f);
#pragma unroll
  for (int i = 0; i < 4; ++i) {
    int c = tid + 256 * i;
    out[(size_t)row * 1024 + c] = (v[i] - mu) * rinv * gamma[c] + beta[c];
  }
}

// ---------------------------------------------------------------------------
// Launch
// ---------------------------------------------------------------------------
extern "C" void kernel_launch(void* const* d_in, const int* in_sizes, int n_in,
                              void* d_out, int out_size, void* d_ws, size_t ws_size,
                              hipStream_t stream) {
  (void)in_sizes; (void)n_in; (void)out_size; (void)ws_size;
  const float* x    = (const float*)d_in[0];
  const float* gWq  = (const float*)d_in[1];
  const float* gbq  = (const float*)d_in[2];
  const float* gWk  = (const float*)d_in[3];
  const float* gbk  = (const float*)d_in[4];
  const float* gWv  = (const float*)d_in[5];
  const float* gbv  = (const float*)d_in[6];
  const float* gWo  = (const float*)d_in[7];
  const float* gbo  = (const float*)d_in[8];
  const float* lWq  = (const float*)d_in[9];
  const float* lbq  = (const float*)d_in[10];
  const float* lWk  = (const float*)d_in[11];
  const float* lbk  = (const float*)d_in[12];
  const float* lWv  = (const float*)d_in[13];
  const float* lbv  = (const float*)d_in[14];
  const float* lWo  = (const float*)d_in[15];
  const float* lbo  = (const float*)d_in[16];
  const float* gamma = (const float*)d_in[17];
  const float* beta  = (const float*)d_in[18];

  char* ws = (char*)d_ws;
  size_t off = 0;
  auto take = [&](size_t bytes) -> void* {
    void* p = ws + off;
    off += (bytes + 255) & ~(size_t)255;
    return p;
  };
  const size_t MAT  = (size_t)MTOT * 1024;   // 8M elements
  const size_t WMAT = (size_t)1024 * 1024;   // 1M elements
  bf16_t* xbf = (bf16_t*)take(MAT * 2);
  bf16_t* wgq = (bf16_t*)take(WMAT * 2);
  bf16_t* wgk = (bf16_t*)take(WMAT * 2);
  bf16_t* wgv = (bf16_t*)take(WMAT * 2);
  bf16_t* wgo = (bf16_t*)take(WMAT * 2);
  bf16_t* wlq = (bf16_t*)take(WMAT * 2);
  bf16_t* wlk = (bf16_t*)take(WMAT * 2);
  bf16_t* wlv = (bf16_t*)take(WMAT * 2);
  bf16_t* wlo = (bf16_t*)take(WMAT * 2);
  bf16_t* qb  = (bf16_t*)take(MAT * 2);
  bf16_t* kb  = (bf16_t*)take(MAT * 2);
  bf16_t* vb  = (bf16_t*)take(MAT * 2);
  bf16_t* og  = (bf16_t*)take(MAT * 2);
  bf16_t* ol  = (bf16_t*)take(MAT * 2);
  float*  yb  = (float*) take(MAT * 4);

  // 1. conversions
  cvt_bf16_kernel<<<2048, 256, 0, stream>>>(x, xbf, (int)MAT);
  cvt_bf16_kernel<<<512, 256, 0, stream>>>(gWq, wgq, (int)WMAT);
  cvt_bf16_kernel<<<512, 256, 0, stream>>>(gWk, wgk, (int)WMAT);
  cvt_bf16_kernel<<<512, 256, 0, stream>>>(gWv, wgv, (int)WMAT);
  cvt_bf16_kernel<<<512, 256, 0, stream>>>(gWo, wgo, (int)WMAT);
  cvt_bf16_kernel<<<512, 256, 0, stream>>>(lWq, wlq, (int)WMAT);
  cvt_bf16_kernel<<<512, 256, 0, stream>>>(lWk, wlk, (int)WMAT);
  cvt_bf16_kernel<<<512, 256, 0, stream>>>(lWv, wlv, (int)WMAT);
  cvt_bf16_kernel<<<512, 256, 0, stream>>>(lWo, wlo, (int)WMAT);

  dim3 gemm_grid(MTOT / 128, 1024 / 64);  // (64, 16)

  // 2. global stream: QKV -> attention
  gemm_qkv_kernel<<<gemm_grid, 128, 0, stream>>>(xbf, wgq, gbq, qb);
  gemm_qkv_kernel<<<gemm_grid, 128, 0, stream>>>(xbf, wgk, gbk, kb);
  gemm_qkv_kernel<<<gemm_grid, 128, 0, stream>>>(xbf, wgv, gbv, vb);
  attn_kernel<<<dim3(SS / 64, HH, BB), 128, 0, stream>>>(qb, kb, vb, og, SS);

  // 3. local stream (reuses qkv buffers; stream ordering serializes)
  gemm_qkv_kernel<<<gemm_grid, 128, 0, stream>>>(xbf, wlq, lbq, qb);
  gemm_qkv_kernel<<<gemm_grid, 128, 0, stream>>>(xbf, wlk, lbk, kb);
  gemm_qkv_kernel<<<gemm_grid, 128, 0, stream>>>(xbf, wlv, lbv, vb);
  attn_kernel<<<dim3(WW / 64, HH, BB * NW), 128, 0, stream>>>(qb, kb, vb, ol, WW);

  // 4. fused output projections + residual, then layernorm
  fuse_proj_kernel<<<gemm_grid, 128, 0, stream>>>(og, wgo, ol, wlo, x, gbo, lbo, yb);
  layernorm_kernel<<<MTOT, 256, 0, stream>>>(yb, gamma, beta, (float*)d_out);
}
